// VectorQuantizer_79010218377603
// MI455X (gfx1250) — matrix operand order repaired and checked by reading, not compiled, not run on hardware
//
#include <hip/hip_runtime.h>
#include <hip/hip_bf16.h>
#include <stdint.h>

// ---------------------------------------------------------------------------
// VQ-VAE vector quantizer for MI455X (gfx1250, wave32, WMMA + TDM)
//   z:        [32, 256, 32, 32] f32
//   codebook: [8192, 256] f32
//   out:      [ z_q (32*256*32*32 f32) | loss (1 f32) | idx (32768 f32) ]
// ---------------------------------------------------------------------------

typedef __attribute__((ext_vector_type(16))) __bf16 v16bf;
typedef __attribute__((ext_vector_type(8)))  __bf16 v8bf;
typedef __attribute__((ext_vector_type(8)))  float  v8f;
typedef __attribute__((ext_vector_type(4)))  unsigned int v4u;
typedef __attribute__((ext_vector_type(8)))  int v8i;
typedef __attribute__((ext_vector_type(4)))  int v4i;

#define DIMD  256
#define KCB   8192
#define BB    32
#define HH    32
#define WW    32
#define NTOT  (BB*HH*WW)     // 32768 latent vectors
#define NELEM (NTOT*DIMD)    // 8388608 elements

__device__ __forceinline__ unsigned short f2bf(float f) {
  union { float f; unsigned int u; } x; x.f = f;
  unsigned int u = x.u;
  u += 0x7fffu + ((u >> 16) & 1u);   // round to nearest even
  return (unsigned short)(u >> 16);
}

// ---------------------------------------------------------------------------
// Kernel 0: [B,C,H,W] f32 -> [N,D] bf16 (LDS 32x32 tile transpose)
// ---------------------------------------------------------------------------
__global__ __launch_bounds__(256)
void vq_transpose_bf16(const float* __restrict__ z, unsigned short* __restrict__ zf) {
  __shared__ float tile[32][33];          // +1 pad: no bank conflicts
  int blk  = blockIdx.x;                  // (b*32 + h)*8 + c_tile
  int ct   = blk & 7;
  int bh   = blk >> 3;
  int h    = bh & 31;
  int b    = bh >> 5;
  int lane = threadIdx.x & 31;
  int sub  = threadIdx.x >> 5;            // 0..7
  int c0   = ct * 32;
  const float* zb = z + (size_t)b * DIMD * HH * WW + (size_t)h * WW;
#pragma unroll
  for (int i = 0; i < 4; ++i) {           // read coalesced along w
    int cl = sub + i * 8;
    tile[cl][lane] = zb[(size_t)(c0 + cl) * HH * WW + lane];
  }
  __syncthreads();
#pragma unroll
  for (int i = 0; i < 4; ++i) {           // write contiguous along c
    int wl = sub + i * 8;
    int n  = (b * HH + h) * WW + wl;
    zf[(size_t)n * DIMD + c0 + lane] = f2bf(tile[lane][wl]);
  }
}

// ---------------------------------------------------------------------------
// Kernel 1: codebook f32 -> bf16, per-row ||c||^2; zero the loss slot
// ---------------------------------------------------------------------------
__global__ __launch_bounds__(256)
void vq_cb_prep(const float* __restrict__ cb, unsigned short* __restrict__ cbbf,
                float* __restrict__ cnorm, float* __restrict__ loss_slot) {
  if (blockIdx.x == 0 && threadIdx.x == 0) *loss_slot = 0.f;
  int wave = blockIdx.x * 8 + (threadIdx.x >> 5);   // one wave32 per row
  int lane = threadIdx.x & 31;
  if (wave >= KCB) return;
  const float* row = cb + (size_t)wave * DIMD;
  unsigned short* orow = cbbf + (size_t)wave * DIMD;
  float s = 0.f;
#pragma unroll
  for (int j = 0; j < DIMD / 32; ++j) {
    float v = row[lane + j * 32];
    orow[lane + j * 32] = f2bf(v);
    s += v * v;
  }
#pragma unroll
  for (int off = 16; off > 0; off >>= 1) s += __shfl_xor(s, off, 32);
  if (lane == 0) cnorm[wave] = s;
}

// ---------------------------------------------------------------------------
// Kernel 2: fused GEMM + row-argmin via v_wmma_f32_16x16x32_bf16 + TDM
//
//   block = 8 waves covering 256 rows; each wave owns TWO 16-row A tiles
//   (full K=256 register-resident).  Codebook columns stream through a
//   double-buffered LDS tile (32 rows x 256 bf16).  The Tensor Data Mover
//   DMAs each 2D tile global->LDS asynchronously (TENSORcnt), with hardware
//   LDS padding: pad_interval=128 dwords (512B row), pad_amount=4 dwords
//   (16B) -> padded row stride 264 bf16, so full-wave ds_load_b128 hits
//   every bank exactly twice (the b128 floor).  Each B fragment read from
//   LDS feeds 2 independent WMMA chains (register-level B reuse).
// ---------------------------------------------------------------------------
#define TCOLS 32                 // codebook columns per LDS tile
#define LSTR  264                // padded LDS row stride in bf16 (528 B)
#define NTILE (KCB / TCOLS)      // 256 tiles

#if __has_builtin(__builtin_amdgcn_tensor_load_to_lds)
#define USE_TDM 1
#else
#define USE_TDM 0
#endif

#if USE_TDM
// Issue one TDM 2D tile load: 32 rows x 512B from gsrc into LDS at lds_off,
// padding +16B after every 512B row.  D# layout per CDNA5 ISA ch.8.
// clang-23 builtin form: (v4u g0, v8i g1, v4i, v4i, v8i, i32 cpol).
__device__ __forceinline__ void tdm_load_tile(const void* gsrc, unsigned lds_off) {
  unsigned long long ga = (unsigned long long)(uintptr_t)gsrc;
  v4u g0;
  g0.x = 1u;                                        // count=1, user mode
  g0.y = lds_off;                                   // lds_addr (bytes)
  g0.z = (unsigned)(ga & 0xffffffffu);              // global_addr[31:0]
  g0.w = (unsigned)((ga >> 32) & 0x1ffffffu)        // global_addr[56:32]
       | (2u << 30);                                // type=2 ("image")
  v8i g1;
  g1[0] = (int)((1u << 16)       // data_size: 1 -> 2-byte elements
              | (1u << 20)       // pad_enable
              | (6u << 22)       // pad_interval: 128 dwords (512 B)
              | (3u << 25));     // pad_amount:   4 dwords  (16 B)
  g1[1] = (int)(256u << 16);     // tensor_dim0 = 256 elems (bits 79:48 lo)
  g1[2] = (int)(8192u << 16);    // tensor_dim0 hi=0 | tensor_dim1 lo = 8192
  g1[3] = (int)(256u << 16);     // tensor_dim1 hi=0 | tile_dim0 = 256
  g1[4] = (int)(TCOLS);          // tile_dim1 = 32 rows, tile_dim2 = 0
  g1[5] = (int)(256);            // tensor_dim0_stride = 256 elems
  g1[6] = 0;                     // stride0 hi | stride1 lo (2D: unused)
  g1[7] = 0;
  v4i z4 = {0, 0, 0, 0};         // groups 2/3: unused for 2D tiles
  v8i z8 = {0, 0, 0, 0, 0, 0, 0, 0};
  __builtin_amdgcn_tensor_load_to_lds(g0, g1, z4, z4, z8, 0);
}
#endif

__global__ __launch_bounds__(256)
void vq_argmin_wmma(const __bf16* __restrict__ zf, const __bf16* __restrict__ cbbf,
                    const float* __restrict__ cnorm, int* __restrict__ idx_i,
                    float* __restrict__ idx_f) {
  __shared__ __align__(16) unsigned short lbuf[2][TCOLS * LSTR];
  __shared__ float lcn[2][TCOLS];

  const int tid     = threadIdx.x;
  const int lane    = tid & 31;
  const int wave    = tid >> 5;
  const int laneN   = lane & 15;          // N coordinate of this lane
  const int half    = lane >> 4;          // lane half selects K sub-chunk
  const int rowbase = blockIdx.x * 256 + wave * 32;

  // ---- A fragments: two 16-row tiles, full K=256, resident in VGPRs ----
  // A 16x32 bf16 layout: both lane halves hold rows M=0..15; per lane two
  // contiguous 8-elt K chunks at offsets c0 and c0+16.
  union AB { v16bf v; v8bf h[2]; };
  AB afrag[2][8];
  const int c0 = half * 8;
#pragma unroll
  for (int rt = 0; rt < 2; ++rt) {
    const __bf16* arow = zf + (size_t)(rowbase + rt * 16 + laneN) * DIMD;
#pragma unroll
    for (int ks = 0; ks < 8; ++ks) {
      afrag[rt][ks].h[0] = *reinterpret_cast<const v8bf*>(arow + ks * 32 + c0);
      afrag[rt][ks].h[1] = *reinterpret_cast<const v8bf*>(arow + ks * 32 + c0 + 16);
    }
  }

  float minv[2][8];
  int   mini[2][8];
#pragma unroll
  for (int rt = 0; rt < 2; ++rt)
#pragma unroll
    for (int r = 0; r < 8; ++r) { minv[rt][r] = 3.4e38f; mini[rt][r] = 0; }

  // ---- prologue: stage tile 0 ----
#if USE_TDM
  if (wave == 0) {
    tdm_load_tile(cbbf, (unsigned)(uintptr_t)&lbuf[0][0]);
    __builtin_amdgcn_s_wait_tensorcnt(0);
  }
#else
#pragma unroll
  for (int j = 0; j < 4; ++j) {
    int p = tid + j * 256;                // 0..1023 float4s
    int row = p >> 5, c4 = p & 31;        // 32 float4 per 512B row
    float4 v = *reinterpret_cast<const float4*>(cbbf + (size_t)row * DIMD + c4 * 8);
    *reinterpret_cast<float4*>(&lbuf[0][row * LSTR + c4 * 8]) = v;
  }
#endif
  if (tid < TCOLS) lcn[0][tid] = cnorm[tid];
  __syncthreads();

  for (int t = 0; t < NTILE; ++t) {
    const int cur = t & 1, nxt = cur ^ 1;
    const bool has_next = (t + 1 < NTILE);

    // ---- kick off next tile's transfer (overlaps this tile's WMMAs) ----
#if USE_TDM
    if (has_next && wave == 0)
      tdm_load_tile(cbbf + (size_t)(t + 1) * TCOLS * DIMD,
                    (unsigned)(uintptr_t)&lbuf[nxt][0]);
    float cntmp = 0.f;
    if (has_next && tid < TCOLS) cntmp = cnorm[(t + 1) * TCOLS + tid];
#else
    float4 tmp[4];
    float  cntmp = 0.f;
    if (has_next) {
      const __bf16* src = cbbf + (size_t)(t + 1) * TCOLS * DIMD;
#pragma unroll
      for (int j = 0; j < 4; ++j) {
        int p = tid + j * 256;
        int row = p >> 5, c4 = p & 31;
        tmp[j] = *reinterpret_cast<const float4*>(src + (size_t)row * DIMD + c4 * 8);
      }
      if (tid < TCOLS) cntmp = cnorm[(t + 1) * TCOLS + tid];
    }
#endif
    if (t + 2 < NTILE)   // warm L2 for t+2 (global_prefetch_b8)
      __builtin_prefetch(cbbf + (size_t)(t + 2) * TCOLS * DIMD + tid * 32, 0, 3);

    // ---- compute on current LDS buffer: 2 subtiles x 8 k-steps x 2 rows ----
#pragma unroll
    for (int s = 0; s < 2; ++s) {
      const unsigned short* bbase = &lbuf[cur][(s * 16 + laneN) * LSTR + half * 8];
      v8f acc0 = {}, acc1 = {};
#pragma unroll
      for (int ks = 0; ks < 8; ++ks) {
        AB bf;
        bf.h[0] = *reinterpret_cast<const v8bf*>(bbase + ks * 32);
        bf.h[1] = *reinterpret_cast<const v8bf*>(bbase + ks * 32 + 16);
        acc0 = __builtin_amdgcn_wmma_f32_16x16x32_bf16(
            false, afrag[0][ks].v, false, bf.v, (short)0, acc0, false, false);
        acc1 = __builtin_amdgcn_wmma_f32_16x16x32_bf16(
            false, afrag[1][ks].v, false, bf.v, (short)0, acc1, false, false);
      }
      const float cn   = lcn[cur][s * 16 + laneN];
      const int   cand = t * TCOLS + s * 16 + laneN;
#pragma unroll
      for (int r = 0; r < 8; ++r) {
        float d0 = cn - 2.0f * acc0[r];   // ||c||^2 - 2 z.c
        float d1 = cn - 2.0f * acc1[r];
        if (d0 < minv[0][r]) { minv[0][r] = d0; mini[0][r] = cand; }
        if (d1 < minv[1][r]) { minv[1][r] = d1; mini[1][r] = cand; }
      }
    }

    // ---- make next tile visible before the barrier ----
    if (has_next) {
#if USE_TDM
      if (wave == 0) __builtin_amdgcn_s_wait_tensorcnt(0);
#else
#pragma unroll
      for (int j = 0; j < 4; ++j) {
        int p = tid + j * 256;
        int row = p >> 5, c4 = p & 31;
        *reinterpret_cast<float4*>(&lbuf[nxt][row * LSTR + c4 * 8]) = tmp[j];
      }
#endif
      if (tid < TCOLS) lcn[nxt][tid] = cntmp;
    }
    __syncthreads();
  }

  // C/D layout: lane = N, VGPR r = row M=r (lanes 0-15) / M=r+8 (lanes 16-31).
  // Reduce (min, argmin) across the 16 lanes sharing each row; first-index ties.
#pragma unroll
  for (int rt = 0; rt < 2; ++rt) {
#pragma unroll
    for (int r = 0; r < 8; ++r) {
      float v = minv[rt][r]; int i = mini[rt][r];
#pragma unroll
      for (int off = 1; off < 16; off <<= 1) {
        float ov = __shfl_xor(v, off, 16);
        int   oi = __shfl_xor(i, off, 16);
        if (ov < v || (ov == v && oi < i)) { v = ov; i = oi; }
      }
      if (laneN == 0) {
        int row = rowbase + rt * 16 + r + half * 8;
        idx_i[row] = i;
        idx_f[row] = (float)i;
      }
    }
  }
}

// ---------------------------------------------------------------------------
// Kernel 3: gather z_q = codebook[idx] back to [B,C,H,W] + fused loss
//   loss = 1.25 * mean((z_q - z)^2)   (stop-gradients collapse in forward)
// ---------------------------------------------------------------------------
__global__ __launch_bounds__(256)
void vq_gather_out(const float* __restrict__ z, const float* __restrict__ cb,
                   const int* __restrict__ idx, float* __restrict__ out,
                   float* __restrict__ loss) {
  __shared__ float red[256];
  size_t g = (size_t)blockIdx.x * 256 + threadIdx.x;   // linear over [B,C,H,W]
  int w = (int)(g & 31);
  int h = (int)((g >> 5) & 31);
  int c = (int)((g >> 10) & 255);
  int b = (int)(g >> 18);
  int n = (b * HH + h) * WW + w;
  int k = idx[n];
  float zq = cb[(size_t)k * DIMD + c];   // f32 codebook, L2-resident gather
  float zv = z[g];
  out[g] = zq;                            // straight-through forward == z_q
  float d = zq - zv;
  red[threadIdx.x] = d * d;
  __syncthreads();
#pragma unroll
  for (int s = 128; s > 0; s >>= 1) {
    if ((int)threadIdx.x < s) red[threadIdx.x] += red[threadIdx.x + s];
    __syncthreads();
  }
  if (threadIdx.x == 0)
    atomicAdd(loss, red[0] * (1.25f / (float)NELEM));
}

// ---------------------------------------------------------------------------
extern "C" void kernel_launch(void* const* d_in, const int* in_sizes, int n_in,
                              void* d_out, int out_size, void* d_ws, size_t ws_size,
                              hipStream_t stream) {
  const float* z  = (const float*)d_in[0];   // [32,256,32,32]
  const float* cb = (const float*)d_in[1];   // [8192,256]

  // workspace layout (all offsets 256B-aligned):
  char* ws = (char*)d_ws;
  size_t off = 0;
  unsigned short* zf   = (unsigned short*)(ws + off); off += (size_t)NTOT * DIMD * 2; // 16 MB
  unsigned short* cbbf = (unsigned short*)(ws + off); off += (size_t)KCB  * DIMD * 2; //  4 MB
  float* cnorm         = (float*)(ws + off);          off += (size_t)KCB * 4;
  int*   idx_i         = (int*)(ws + off);            off += (size_t)NTOT * 4;

  float* out   = (float*)d_out;
  float* loss  = out + NELEM;        // scalar
  float* idx_f = out + NELEM + 1;    // 32768 floats

  vq_transpose_bf16<<<BB * HH * (DIMD / 32), 256, 0, stream>>>(z, zf);
  vq_cb_prep<<<KCB / 8, 256, 0, stream>>>(cb, cbbf, cnorm, loss);
  vq_argmin_wmma<<<NTOT / 256, 256, 0, stream>>>(
      (const __bf16*)zf, (const __bf16*)cbbf, cnorm, idx_i, idx_f);
  vq_gather_out<<<NELEM / 256, 256, 0, stream>>>(z, cb, idx_i, out, loss);
}
